// HierarchicalLstmDecoder_20315195310550
// MI455X (gfx1250) — compile-verified, matrix operand order checked by reading, not run on hardware
//
#include <hip/hip_runtime.h>
#include <hip/hip_bf16.h>
#include <cstdint>
#include <cstddef>

typedef __bf16 bf16_t;
typedef __attribute__((ext_vector_type(16))) __bf16 v16bf;
typedef __attribute__((ext_vector_type(8)))  float  v8f;

// ---- problem dims (from reference) ----
constexpr int cB = 128, cS = 16, cL = 16, cT = 256;
constexpr int cZ = 512, cHC = 1024, cE = 512;
constexpr int cHD = 1024, cV = 1024;

// ---- GEMM tile config: 128x128 block, K-step 32 (one WMMA K), 8 waves ----
constexpr int BM = 128, BN = 128, BK = 32, NTHREADS = 256;

enum { EP_TANH = 1, EP_OUT_BF16 = 2, EP_ADDC = 4 };

// CDNA5 async copy: global -> LDS without VGPR staging, tracked by ASYNCcnt.
// lds_off is the raw byte offset into the workgroup LDS allocation (single
// __shared__ array => allocation starts at LDS offset 0).
// IMPORTANT: the LDS object's pointer is passed as a (unused) operand so the
// array escapes into the asm — otherwise LLVM alias analysis assumes the LDS
// is never written and deletes the ds_load fragment reads (observed round 2).
__device__ __forceinline__ void async_ld_b128(unsigned lds_off, const void* gaddr,
                                              const void* lds_escape)
{
  asm volatile("global_load_async_to_lds_b128 %0, %1, off"
               :: "v"(lds_off), "v"(gaddr), "v"(lds_escape)
               : "memory");
}
__device__ __forceinline__ void wait_async0()
{
  asm volatile("s_wait_asynccnt 0x0" ::: "memory");
}

// C[M,N] = A[M,K](bf16) x W[N,K]^T(bf16) (+bias) with optional epilogue.
// Row permute on store: outr = (r % p1) * p0 + r / p1 when p1 > 0.
__global__ __launch_bounds__(NTHREADS)
void k_gemm_bf16(const bf16_t* __restrict__ A,
                 const bf16_t* __restrict__ Wm,
                 float*  __restrict__ Cf,
                 bf16_t* __restrict__ Cb,
                 const float* __restrict__ bias,
                 int M, int N, int K,
                 int flags, int p0, int p1)
{
  // [buf][A=0/B=1][row][col] — single shared object so LDS offsets start at 0.
  __shared__ __align__(16) bf16_t sm[2][2][BM][BK];
  void* smbase = (void*)&sm[0][0][0][0];

  const int tid  = threadIdx.x;
  const int lane = tid & 31;
  const int wave = tid >> 5;      // 8 waves
  const int wm   = wave >> 2;     // 0..1 : 4 m-tiles each
  const int wn   = wave & 3;      // 0..3 : 2 n-tiles each
  const int l16  = lane & 15;
  const int half = lane >> 4;

  const int bm = blockIdx.y * BM;
  const int bn = blockIdx.x * BN;

  const v8f z8 = {0.f,0.f,0.f,0.f,0.f,0.f,0.f,0.f};
  v8f acc[4][2];
  #pragma unroll
  for (int mt = 0; mt < 4; ++mt)
    #pragma unroll
    for (int nt = 0; nt < 2; ++nt) acc[mt][nt] = z8;

  const int nk = K / BK;

  // Per-thread fill: tile is 128 rows x 64 B = 512 x 16B chunks; 2 per thread.
  const int c0 = tid,            r0 = c0 >> 2, o0 = (c0 & 3) * 8;
  const int c1 = tid + NTHREADS, r1 = c1 >> 2, o1 = (c1 & 3) * 8;
  // LDS byte offsets (sm element offset * 2 bytes)
  auto lds_off = [&](int buf, int ab, int r, int o) -> unsigned {
    return (unsigned)((((buf * 2 + ab) * BM + r) * BK + o) * 2);
  };

  // preload k-tile 0 into buffer 0 (async DMA to LDS)
  {
    const bf16_t* ga = A  + (size_t)bm * K;
    const bf16_t* gw = Wm + (size_t)bn * K;
    async_ld_b128(lds_off(0, 0, r0, o0), ga + (size_t)r0 * K + o0, smbase);
    async_ld_b128(lds_off(0, 1, r0, o0), gw + (size_t)r0 * K + o0, smbase);
    async_ld_b128(lds_off(0, 0, r1, o1), ga + (size_t)r1 * K + o1, smbase);
    async_ld_b128(lds_off(0, 1, r1, o1), gw + (size_t)r1 * K + o1, smbase);
  }

  for (int kt = 0; kt < nk; ++kt) {
    const int buf = kt & 1;
    wait_async0();     // my async fills for tile kt have landed in LDS
    __syncthreads();   // everyone's fills landed; everyone done reading buf^1
    if (kt + 1 < nk) { // overlap next-tile DMA with this tile's WMMAs
      const bf16_t* ga = A  + (size_t)bm * K + (size_t)(kt + 1) * BK;
      const bf16_t* gw = Wm + (size_t)bn * K + (size_t)(kt + 1) * BK;
      async_ld_b128(lds_off(buf ^ 1, 0, r0, o0), ga + (size_t)r0 * K + o0, smbase);
      async_ld_b128(lds_off(buf ^ 1, 1, r0, o0), gw + (size_t)r0 * K + o0, smbase);
      async_ld_b128(lds_off(buf ^ 1, 0, r1, o1), ga + (size_t)r1 * K + o1, smbase);
      async_ld_b128(lds_off(buf ^ 1, 1, r1, o1), gw + (size_t)r1 * K + o1, smbase);
      if (kt + 2 < nk) { // warm GL2 for the tile after that
        __builtin_prefetch(ga + (size_t)r0 * K + o0 + BK, 0, 1);
        __builtin_prefetch(gw + (size_t)r0 * K + o0 + BK, 0, 1);
      }
    }

    // A fragments: lane l16 = row-in-tile, half selects K-halves (8 bf16 each)
    v16bf af[4];
    #pragma unroll
    for (int mt = 0; mt < 4; ++mt) {
      const bf16_t* p = &sm[buf][0][(wm * 4 + mt) * 16 + l16][0];
      union { uint4 u[2]; v16bf v; } f;
      f.u[0] = *(const uint4*)(p + half * 8);
      f.u[1] = *(const uint4*)(p + 16 + half * 8);
      af[mt] = f.v;
    }
    #pragma unroll
    for (int nt = 0; nt < 2; ++nt) {
      const bf16_t* p = &sm[buf][1][(wn * 2 + nt) * 16 + l16][0];
      union { uint4 u[2]; v16bf v; } f;
      f.u[0] = *(const uint4*)(p + half * 8);
      f.u[1] = *(const uint4*)(p + 16 + half * 8);
      v16bf wf = f.v;
      #pragma unroll
      for (int mt = 0; mt < 4; ++mt)
        acc[mt][nt] = __builtin_amdgcn_wmma_f32_16x16x32_bf16(
            false, af[mt], false, wf, (short)0, acc[mt][nt], false, false);
    }
  }

  // epilogue: C/D layout — VGPR v: row-in-tile = half*8+v, col-in-tile = l16
  #pragma unroll
  for (int mt = 0; mt < 4; ++mt) {
    #pragma unroll
    for (int nt = 0; nt < 2; ++nt) {
      const int colg = bn + (wn * 2 + nt) * 16 + l16;
      const float bv = bias ? bias[colg] : 0.f;
      #pragma unroll
      for (int v = 0; v < 8; ++v) {
        const int rowg = bm + (wm * 4 + mt) * 16 + half * 8 + v;
        float val = acc[mt][nt][v] + bv;
        if (flags & EP_ADDC) val += Cf[(size_t)rowg * N + colg];
        if (flags & EP_TANH) val = tanhf(val);
        int outr = rowg;
        if (p1 > 0) outr = (rowg % p1) * p0 + rowg / p1;
        if (flags & EP_OUT_BF16) Cb[(size_t)outr * N + colg] = (bf16_t)val;
        else                     Cf[(size_t)outr * N + colg] = val;
      }
    }
  }
}

// LSTM gate pointwise: g holds [i f g o] (biases already folded in), c in/out f32,
// h written bf16 (feeds next-step recurrent WMMA GEMM directly).
__global__ void k_lstm_point(const float* __restrict__ g, float* __restrict__ c,
                             bf16_t* __restrict__ h, int Bn, int H)
{
  int idx = blockIdx.x * blockDim.x + threadIdx.x;
  if (idx >= Bn * H) return;
  int b = idx / H, j = idx - b * H;
  const float* gr = g + (size_t)b * 4 * H;
  float xi = gr[j], xf = gr[H + j], xg = gr[2 * H + j], xo = gr[3 * H + j];
  float si = 1.f / (1.f + __expf(-xi));
  float sf = 1.f / (1.f + __expf(-xf));
  float so = 1.f / (1.f + __expf(-xo));
  float cn = sf * c[idx] + si * tanhf(xg);
  c[idx] = cn;
  h[idx] = (bf16_t)(so * tanhf(cn));
}

// dec_in[t*B+b, :] = [ prev_token(b,t) (V) | emb(b, t/L) (E) ]  in bf16
__global__ void k_build_decin(const float* __restrict__ x,    // [B,T,V]
                              const bf16_t* __restrict__ emb, // [B,S,E]
                              bf16_t* __restrict__ di)        // [T*B, V+E]
{
  const long long total = (long long)cT * cB * (cV + cE);
  long long idx = (long long)blockIdx.x * blockDim.x + threadIdx.x;
  if (idx >= total) return;
  int col = (int)(idx % (cV + cE));
  int r   = (int)(idx / (cV + cE));
  int b = r % cB, t = r / cB;
  bf16_t val;
  if (col < cV) {
    float f = (t == 0) ? 0.f : x[((size_t)b * cT + (t - 1)) * cV + col];
    val = (bf16_t)f;
  } else {
    val = emb[((size_t)b * cS + (t / cL)) * cE + (col - cV)];
  }
  di[idx] = val;
}

__global__ void k_f32_to_bf16(const float* __restrict__ s, bf16_t* __restrict__ d, size_t n)
{
  size_t i = (size_t)blockIdx.x * blockDim.x + threadIdx.x;
  if (i < n) d[i] = (bf16_t)s[i];
}

__global__ void k_zero_f32(float* p, size_t n)
{
  size_t i = (size_t)blockIdx.x * blockDim.x + threadIdx.x;
  if (i < n) p[i] = 0.f;
}

__global__ void k_zero_bf16(bf16_t* p, size_t n)
{
  size_t i = (size_t)blockIdx.x * blockDim.x + threadIdx.x;
  if (i < n) p[i] = (bf16_t)0.f;
}

__global__ void k_add_bias(const float* __restrict__ a, const float* __restrict__ b,
                           float* __restrict__ o, int n)
{
  int i = blockIdx.x * blockDim.x + threadIdx.x;
  if (i < n) o[i] = a[i] + b[i];
}

extern "C" void kernel_launch(void* const* d_in, const int* in_sizes, int n_in,
                              void* d_out, int out_size, void* d_ws, size_t ws_size,
                              hipStream_t stream)
{
  (void)in_sizes; (void)n_in; (void)out_size; (void)ws_size;
  const float* z      = (const float*)d_in[0];
  const float* x      = (const float*)d_in[1];
  const float* fcz_w  = (const float*)d_in[2];
  const float* fcz_b  = (const float*)d_in[3];
  const float* c_wih  = (const float*)d_in[4];
  const float* c_whh  = (const float*)d_in[5];
  const float* c_bih  = (const float*)d_in[6];
  const float* c_bhh  = (const float*)d_in[7];
  const float* fcc_w  = (const float*)d_in[8];
  const float* fcc_b  = (const float*)d_in[9];
  const float* d0_wih = (const float*)d_in[10];
  const float* d0_whh = (const float*)d_in[11];
  const float* d0_bih = (const float*)d_in[12];
  const float* d0_bhh = (const float*)d_in[13];
  const float* d1_wih = (const float*)d_in[14];
  const float* d1_whh = (const float*)d_in[15];
  const float* d1_bih = (const float*)d_in[16];
  const float* d1_bhh = (const float*)d_in[17];
  const float* out_w  = (const float*)d_in[18];
  const float* out_b  = (const float*)d_in[19];
  float* logits = (float*)d_out;

  char* ws = (char*)d_ws;
  size_t off = 0;
  auto alloc = [&](size_t bytes) -> char* {
    char* p = ws + off;
    off = (off + bytes + 255) & ~(size_t)255;
    return p;
  };

  bf16_t* z_bf    = (bf16_t*)alloc((size_t)cB * cS * cZ * 2);
  bf16_t* fcz_wb  = (bf16_t*)alloc((size_t)cHC * cZ * 2);
  bf16_t* c_wihb  = (bf16_t*)alloc((size_t)4 * cHC * cHC * 2);
  bf16_t* c_whhb  = (bf16_t*)alloc((size_t)4 * cHC * cHC * 2);
  bf16_t* fcc_wb  = (bf16_t*)alloc((size_t)cE * cHC * 2);
  bf16_t* d0_wihb = (bf16_t*)alloc((size_t)4 * cHD * (cV + cE) * 2);
  bf16_t* d0_whhb = (bf16_t*)alloc((size_t)4 * cHD * cHD * 2);
  bf16_t* d1_wihb = (bf16_t*)alloc((size_t)4 * cHD * cHD * 2);
  bf16_t* d1_whhb = (bf16_t*)alloc((size_t)4 * cHD * cHD * 2);
  bf16_t* out_wb  = (bf16_t*)alloc((size_t)cV * cHD * 2);
  float*  bias_c  = (float*)alloc((size_t)4 * cHC * 4);
  float*  bias_d0 = (float*)alloc((size_t)4 * cHD * 4);
  float*  bias_d1 = (float*)alloc((size_t)4 * cHD * 4);
  bf16_t* cond_in = (bf16_t*)alloc((size_t)cS * cB * cHC * 2);         // [S,B,HC]
  float*  gx_c    = (float*)alloc((size_t)cS * cB * 4 * cHC * 4);      // [S,B,4HC]
  bf16_t* hs_c    = (bf16_t*)alloc((size_t)(cS + 1) * cB * cHC * 2);   // slot0 = zeros
  float*  c_c     = (float*)alloc((size_t)cB * cHC * 4);
  bf16_t* emb     = (bf16_t*)alloc((size_t)cB * cS * cE * 2);          // [B,S,E]
  bf16_t* dec_in  = (bf16_t*)alloc((size_t)cT * cB * (cV + cE) * 2);   // [T*B,V+E]
  float*  gx      = (float*)alloc((size_t)cT * cB * 4 * cHD * 4);      // reused L0/L1
  bf16_t* h0      = (bf16_t*)alloc((size_t)(cT + 1) * cB * cHD * 2);   // slot0 = zeros
  bf16_t* h1      = (bf16_t*)alloc((size_t)(cT + 1) * cB * cHD * 2);   // slot0 = zeros
  float*  c0      = (float*)alloc((size_t)cB * cHD * 4);
  float*  c1      = (float*)alloc((size_t)cB * cHD * 4);

  auto cvt = [&](const float* s, bf16_t* d, size_t n) {
    k_f32_to_bf16<<<dim3((unsigned)((n + 255) / 256)), dim3(256), 0, stream>>>(s, d, n);
  };
  auto zf = [&](float* p, size_t n) {
    k_zero_f32<<<dim3((unsigned)((n + 255) / 256)), dim3(256), 0, stream>>>(p, n);
  };
  auto zb = [&](bf16_t* p, size_t n) {
    k_zero_bf16<<<dim3((unsigned)((n + 255) / 256)), dim3(256), 0, stream>>>(p, n);
  };
  auto gemm = [&](const bf16_t* A, const bf16_t* Wm, float* Cf, bf16_t* Cb,
                  const float* bias, int M, int N, int K, int flags, int p0, int p1) {
    k_gemm_bf16<<<dim3(N / BN, M / BM), dim3(NTHREADS), 0, stream>>>(
        A, Wm, Cf, Cb, bias, M, N, K, flags, p0, p1);
  };

  // ---- weight / input conversion to bf16 (weights stay resident in 192MB L2) ----
  cvt(z,      z_bf,    (size_t)cB * cS * cZ);
  cvt(fcz_w,  fcz_wb,  (size_t)cHC * cZ);
  cvt(c_wih,  c_wihb,  (size_t)4 * cHC * cHC);
  cvt(c_whh,  c_whhb,  (size_t)4 * cHC * cHC);
  cvt(fcc_w,  fcc_wb,  (size_t)cE * cHC);
  cvt(d0_wih, d0_wihb, (size_t)4 * cHD * (cV + cE));
  cvt(d0_whh, d0_whhb, (size_t)4 * cHD * cHD);
  cvt(d1_wih, d1_wihb, (size_t)4 * cHD * cHD);
  cvt(d1_whh, d1_whhb, (size_t)4 * cHD * cHD);
  cvt(out_w,  out_wb,  (size_t)cV * cHD);
  k_add_bias<<<dim3((4 * cHC + 255) / 256), dim3(256), 0, stream>>>(c_bih, c_bhh, bias_c, 4 * cHC);
  k_add_bias<<<dim3((4 * cHD + 255) / 256), dim3(256), 0, stream>>>(d0_bih, d0_bhh, bias_d0, 4 * cHD);
  k_add_bias<<<dim3((4 * cHD + 255) / 256), dim3(256), 0, stream>>>(d1_bih, d1_bhh, bias_d1, 4 * cHD);

  // ---- zero-init recurrent state every call (deterministic under graph replay) ----
  zb(hs_c, (size_t)cB * cHC);  zf(c_c, (size_t)cB * cHC);
  zb(h0,   (size_t)cB * cHD);  zf(c0,  (size_t)cB * cHD);
  zb(h1,   (size_t)cB * cHD);  zf(c1,  (size_t)cB * cHD);

  // ---- conductor ----
  // cond_in[s,b] = z[b,s] @ fcz_w^T + fcz_b   (permute (b,s)->(s,b), bf16 out)
  gemm(z_bf, fcz_wb, nullptr, cond_in, fcz_b, cB * cS, cHC, cZ, EP_OUT_BF16, cB, cS);
  // gx_c = cond_in @ c_wih^T + (c_bih + c_bhh)   (time-parallel input projection)
  gemm(cond_in, c_wihb, gx_c, nullptr, bias_c, cS * cB, 4 * cHC, cHC, 0, 0, 0);
  for (int s = 0; s < cS; ++s) {
    gemm(hs_c + (size_t)s * cB * cHC, c_whhb, gx_c + (size_t)s * cB * 4 * cHC, nullptr,
         nullptr, cB, 4 * cHC, cHC, EP_ADDC, 0, 0);
    k_lstm_point<<<dim3((cB * cHC + 255) / 256), dim3(256), 0, stream>>>(
        gx_c + (size_t)s * cB * 4 * cHC, c_c, hs_c + (size_t)(s + 1) * cB * cHC, cB, cHC);
  }
  // emb[b,s] = tanh(hs_c[s,b] @ fcc_w^T + fcc_b)   (permute (s,b)->(b,s), bf16 out)
  gemm(hs_c + (size_t)cB * cHC, fcc_wb, nullptr, emb, fcc_b, cS * cB, cE, cHC,
       EP_TANH | EP_OUT_BF16, cS, cB);

  // ---- decoder input assembly + layer-0 input projection ----
  {
    long long total = (long long)cT * cB * (cV + cE);
    k_build_decin<<<dim3((unsigned)((total + 255) / 256)), dim3(256), 0, stream>>>(x, emb, dec_in);
  }
  gemm(dec_in, d0_wihb, gx, nullptr, bias_d0, cT * cB, 4 * cHD, cV + cE, 0, 0, 0);
  for (int t = 0; t < cT; ++t) {
    gemm(h0 + (size_t)t * cB * cHD, d0_whhb, gx + (size_t)t * cB * 4 * cHD, nullptr,
         nullptr, cB, 4 * cHD, cHD, EP_ADDC, 0, 0);
    k_lstm_point<<<dim3((cB * cHD + 255) / 256), dim3(256), 0, stream>>>(
        gx + (size_t)t * cB * 4 * cHD, c0, h0 + (size_t)(t + 1) * cB * cHD, cB, cHD);
  }

  // ---- layer-1 (reuse gx buffer) ----
  gemm(h0 + (size_t)cB * cHD, d1_wihb, gx, nullptr, bias_d1, cT * cB, 4 * cHD, cHD, 0, 0, 0);
  for (int t = 0; t < cT; ++t) {
    gemm(h1 + (size_t)t * cB * cHD, d1_whhb, gx + (size_t)t * cB * 4 * cHD, nullptr,
         nullptr, cB, 4 * cHD, cHD, EP_ADDC, 0, 0);
    k_lstm_point<<<dim3((cB * cHD + 255) / 256), dim3(256), 0, stream>>>(
        gx + (size_t)t * cB * 4 * cHD, c1, h1 + (size_t)(t + 1) * cB * cHD, cB, cHD);
  }

  // ---- output projection, permute (t,b)->(b,t), f32 logits ----
  gemm(h1 + (size_t)cB * cHD, out_wb, logits, nullptr, out_b, cT * cB, cV, cHD, 0, cT, cB);
}
